// SpatialAttention_24111946399968
// MI455X (gfx1250) — matrix operand order repaired and checked by reading, not compiled, not run on hardware
//
#include <hip/hip_runtime.h>
#include <math.h>

#define B_ 16
#define C_ 12
#define N_ 2048
#define F_ 64

typedef __attribute__((ext_vector_type(16))) _Float16 v16h;
typedef __attribute__((ext_vector_type(8)))  _Float16 v8h;
typedef __attribute__((ext_vector_type(8)))  float    v8f;
typedef __attribute__((ext_vector_type(4)))  int      v4i;

// ---------------------------------------------------------------------------
// Step 1: y1a[b,n,f] = sum_c x[b,c,n,f] * w1[c]
// ---------------------------------------------------------------------------
__global__ void reduce_c_kernel(const float* __restrict__ x,
                                const float* __restrict__ w1,
                                float* __restrict__ y1a) {
    int idx = blockIdx.x * 256 + threadIdx.x;          // b*N*F + n*F + f
    int f = idx & (F_ - 1);
    int n = (idx >> 6) & (N_ - 1);
    int b = idx >> 17;
    const float* xp = x + ((size_t)b * C_ * N_ + n) * F_ + f;
    float acc = 0.f;
#pragma unroll
    for (int c = 0; c < C_; ++c)
        acc += xp[(size_t)c * N_ * F_] * w1[c];
    y1a[idx] = acc;
}

// ---------------------------------------------------------------------------
// Step 3: y2[b,c,n] = sum_f x[b,c,n,f] * w3[f]   (one wave32 per output)
// ---------------------------------------------------------------------------
__global__ void reduce_f_kernel(const float* __restrict__ x,
                                const float* __restrict__ w3,
                                float* __restrict__ y2) {
    int lane = threadIdx.x & 31;
    int o = blockIdx.x * 8 + (threadIdx.x >> 5);       // b*C*N + c*N + n
    size_t base = (size_t)o * F_;                      // x is (B,C,N,F) flat
    float acc = x[base + lane] * w3[lane] + x[base + lane + 32] * w3[lane + 32];
#pragma unroll
    for (int off = 16; off > 0; off >>= 1)
        acc += __shfl_xor(acc, off, 32);
    if (lane == 0) y2[o] = acc;
}

// ---------------------------------------------------------------------------
// Step 2: y1[b,n,c] = sum_f y1a[b,n,f] * w2[c,f]
// ---------------------------------------------------------------------------
__global__ void y1_kernel(const float* __restrict__ y1a,
                          const float* __restrict__ w2,
                          float* __restrict__ y1) {
    int idx = blockIdx.x * 256 + threadIdx.x;          // b*N*C + n*C + c
    int c = idx % C_;
    int bn = idx / C_;                                 // b*N + n
    const float* ap = y1a + (size_t)bn * F_;
    const float* wp = w2 + (size_t)c * F_;
    float acc = 0.f;
#pragma unroll
    for (int f = 0; f < F_; ++f) acc += ap[f] * wp[f];
    y1[idx] = acc;
}

// ---------------------------------------------------------------------------
// vs -> f16
// ---------------------------------------------------------------------------
__global__ void cvt_vs_kernel(const float* __restrict__ vs,
                              _Float16* __restrict__ vsh) {
    int idx = blockIdx.x * 256 + threadIdx.x;
    vsh[idx] = (_Float16)vs[idx];
}

// ---------------------------------------------------------------------------
// Steps 4+5: sT[b,q,p] = f16(sigmoid(sum_c y1[b,p,c]*y2[b,c,q] + bs[p,q]))
// Written TRANSPOSED (p contiguous) so GEMM B-fragments load contiguously.
// ---------------------------------------------------------------------------
__global__ void sig_kernel(const float* __restrict__ y1,
                           const float* __restrict__ y2,
                           const float* __restrict__ bs,
                           _Float16* __restrict__ sT) {
    int p = blockIdx.x * 256 + threadIdx.x;
    int q = blockIdx.y;
    int b = blockIdx.z;
    float acc = bs[(size_t)p * N_ + q];
    const float* y1p = y1 + ((size_t)b * N_ + p) * C_;
    const float* y2p = y2 + (size_t)b * C_ * N_ + q;
#pragma unroll
    for (int c = 0; c < C_; ++c)
        acc += y1p[c] * y2p[(size_t)c * N_];
    float s = 1.0f / (1.0f + __expf(-acc));
    sT[((size_t)b * N_ + q) * N_ + p] = (_Float16)s;
}

// ---------------------------------------------------------------------------
// Step 6: out[b,n,q] = sum_p vs[n,p] * s[b,p,q]
// Block tile 128(m) x 256(q), 8 waves in 2x4 grid, wave tile 64x64
// (4x4 wmma accumulators), K-step 32, double-buffered LDS with one barrier
// per K-step, async global->LDS prefetch overlapped with WMMA.
// ---------------------------------------------------------------------------
#define LSTR 40   // LDS row stride in halves (32 data + 8 pad; 80B = 5x16B)

#if __has_builtin(__builtin_amdgcn_global_load_async_to_lds_b128) && \
    __has_builtin(__builtin_amdgcn_s_wait_asynccnt)
#define USE_ASYNC 1
#else
#define USE_ASYNC 0
#endif

#if USE_ASYNC
typedef __attribute__((address_space(1))) v4i* gv4i_p;   // global, 64-bit ptr
typedef __attribute__((address_space(3))) v4i* lv4i_p;   // LDS, 32-bit ptr
#endif

__device__ __forceinline__ void cp16(void* l, const void* g) {
#if USE_ASYNC
    __builtin_amdgcn_global_load_async_to_lds_b128(
        (gv4i_p)(uintptr_t)g,
        (lv4i_p)(uint32_t)(uintptr_t)l,
        0, 0);
#else
    *(v8h*)l = *(const v8h*)g;
#endif
}

__device__ __forceinline__ void wait_stage() {
#if USE_ASYNC
    __builtin_amdgcn_s_wait_asynccnt(0);
#endif
    __syncthreads();
}

__device__ __forceinline__ v16h cat8(v8h lo, v8h hi) {
    v16h r;
#pragma unroll
    for (int i = 0; i < 8; ++i) { r[i] = lo[i]; r[i + 8] = hi[i]; }
    return r;
}

__global__ __launch_bounds__(256, 1) void gemm_kernel(
        const _Float16* __restrict__ Ah,   // vs_h (N x N), row-major (n,p)
        const _Float16* __restrict__ Bt,   // sT (B,N,N), (q,p) p-contiguous
        float* __restrict__ out) {
    __shared__ _Float16 lA[2][128 * LSTR];   // 20 KB
    __shared__ _Float16 lB[2][256 * LSTR];   // 40 KB

    int q0 = blockIdx.x * 256;
    int n0 = blockIdx.y * 128;
    int b  = blockIdx.z;
    int tid  = threadIdx.x;
    int lane = tid & 31;
    int wave = tid >> 5;
    int wm = wave & 1;        // 2 waves along n (64 rows each)
    int wq = wave >> 1;       // 4 waves along q (64 cols each)

    const _Float16* Bb = Bt + (size_t)b * N_ * N_;

    v8f acc[4][4];
#pragma unroll
    for (int i = 0; i < 4; ++i)
#pragma unroll
        for (int j = 0; j < 4; ++j)
#pragma unroll
            for (int e = 0; e < 8; ++e) acc[i][j][e] = 0.f;

    // staging geometry: 16B chunks, 4 per 32-half row
    int sr = tid >> 2;            // 0..63
    int sc = (tid & 3) * 8;       // 0,8,16,24

    // fragment geometry (16-bit WMMA A/B lane layouts)
    int arow = wm * 64 + (lane & 15);
    int c0   = (lane < 16) ? 0 : 8;     // A: K chunks [c0..c0+7],[c0+16..c0+23]
    int brow = wq * 64 + (lane & 15);
    int kb   = (lane < 16) ? 0 : 8;     // B: K chunks [kb..kb+7],[kb+16..kb+23]

#define STAGE(buf, p0)                                                        \
    do {                                                                      \
        cp16(&lA[buf][(size_t)(sr) * LSTR + sc],                              \
             &Ah[(size_t)(n0 + sr) * N_ + (p0) + sc]);                        \
        cp16(&lA[buf][(size_t)(sr + 64) * LSTR + sc],                         \
             &Ah[(size_t)(n0 + sr + 64) * N_ + (p0) + sc]);                   \
        cp16(&lB[buf][(size_t)(sr) * LSTR + sc],                              \
             &Bb[(size_t)(q0 + sr) * N_ + (p0) + sc]);                        \
        cp16(&lB[buf][(size_t)(sr + 64) * LSTR + sc],                         \
             &Bb[(size_t)(q0 + sr + 64) * N_ + (p0) + sc]);                   \
        cp16(&lB[buf][(size_t)(sr + 128) * LSTR + sc],                        \
             &Bb[(size_t)(q0 + sr + 128) * N_ + (p0) + sc]);                  \
        cp16(&lB[buf][(size_t)(sr + 192) * LSTR + sc],                        \
             &Bb[(size_t)(q0 + sr + 192) * N_ + (p0) + sc]);                  \
    } while (0)

    STAGE(0, 0);

    for (int i = 0; i < N_ / 32; ++i) {
        int cur = i & 1;
        wait_stage();                      // cur buffer resident, prev reads done
        if (i + 1 < N_ / 32)
            STAGE(cur ^ 1, (i + 1) * 32);  // prefetch overlaps the WMMAs below

        v16h afr[4];
#pragma unroll
        for (int tm = 0; tm < 4; ++tm) {
            const _Float16* ap = &lA[cur][(size_t)(arow + tm * 16) * LSTR];
            afr[tm] = cat8(*(const v8h*)&ap[c0], *(const v8h*)&ap[c0 + 16]);
        }
        v16h bfr[4];
#pragma unroll
        for (int tq = 0; tq < 4; ++tq) {
            const _Float16* bp = &lB[cur][(size_t)(brow + tq * 16) * LSTR + kb];
            bfr[tq] = cat8(*(const v8h*)&bp[0], *(const v8h*)&bp[16]);
        }

#pragma unroll
        for (int tm = 0; tm < 4; ++tm)
#pragma unroll
            for (int tq = 0; tq < 4; ++tq)
                acc[tm][tq] = __builtin_amdgcn_wmma_f32_16x16x32_f16(
                    false, afr[tm], false, bfr[tq],
                    (short)0, acc[tm][tq], false, false);
    }
#undef STAGE

    // epilogue: D layout -> lanes 0-15: M=r, N=lane ; lanes 16-31: M=8+r
    int colq  = q0 + wq * 64 + (lane & 15);
    int rbase = n0 + wm * 64 + ((lane >> 4) << 3);
#pragma unroll
    for (int tm = 0; tm < 4; ++tm)
#pragma unroll
        for (int tq = 0; tq < 4; ++tq)
#pragma unroll
            for (int r = 0; r < 8; ++r)
                out[((size_t)b * N_ + rbase + tm * 16 + r) * N_ + colq + tq * 16] =
                    acc[tm][tq][r];
}

// ---------------------------------------------------------------------------
// Step 7: in-place row softmax over last axis (rows of 2048)
// ---------------------------------------------------------------------------
__global__ void softmax_kernel(float* __restrict__ y) {
    __shared__ float redm[8];
    __shared__ float reds[8];
    float* p = y + (size_t)blockIdx.x * N_;
    int lane = threadIdx.x & 31;
    int wave = threadIdx.x >> 5;

    float v[8];
    float vmax = -INFINITY;
#pragma unroll
    for (int i = 0; i < 8; ++i) {
        v[i] = p[threadIdx.x + i * 256];
        vmax = fmaxf(vmax, v[i]);
    }
#pragma unroll
    for (int off = 16; off > 0; off >>= 1)
        vmax = fmaxf(vmax, __shfl_xor(vmax, off, 32));
    if (lane == 0) redm[wave] = vmax;
    __syncthreads();
    float m = redm[0];
#pragma unroll
    for (int w = 1; w < 8; ++w) m = fmaxf(m, redm[w]);

    float s = 0.f;
#pragma unroll
    for (int i = 0; i < 8; ++i) { v[i] = __expf(v[i] - m); s += v[i]; }
#pragma unroll
    for (int off = 16; off > 0; off >>= 1)
        s += __shfl_xor(s, off, 32);
    if (lane == 0) reds[wave] = s;
    __syncthreads();
    float tot = 0.f;
#pragma unroll
    for (int w = 0; w < 8; ++w) tot += reds[w];
    float inv = 1.0f / tot;
#pragma unroll
    for (int i = 0; i < 8; ++i)
        p[threadIdx.x + i * 256] = v[i] * inv;
}

// ---------------------------------------------------------------------------
extern "C" void kernel_launch(void* const* d_in, const int* in_sizes, int n_in,
                              void* d_out, int out_size, void* d_ws, size_t ws_size,
                              hipStream_t stream) {
    (void)in_sizes; (void)n_in; (void)out_size; (void)ws_size;
    const float* x  = (const float*)d_in[0];
    const float* w1 = (const float*)d_in[1];
    const float* w2 = (const float*)d_in[2];
    const float* w3 = (const float*)d_in[3];
    const float* vs = (const float*)d_in[4];
    const float* bs = (const float*)d_in[5];
    float* out = (float*)d_out;

    // workspace layout (~154 MB)
    float* y1a = (float*)d_ws;                                  // B*N*F
    float* y2  = y1a + (size_t)B_ * N_ * F_;                    // B*C*N
    float* y1  = y2  + (size_t)B_ * C_ * N_;                    // B*N*C
    _Float16* vsh = (_Float16*)(y1 + (size_t)B_ * N_ * C_);     // N*N
    _Float16* sT  = vsh + (size_t)N_ * N_;                      // B*N*N

    reduce_c_kernel<<<(B_ * N_ * F_) / 256, 256, 0, stream>>>(x, w1, y1a);
    reduce_f_kernel<<<(B_ * C_ * N_) / 8, 256, 0, stream>>>(x, w3, y2);
    y1_kernel<<<(B_ * N_ * C_) / 256, 256, 0, stream>>>(y1a, w2, y1);
    cvt_vs_kernel<<<((size_t)N_ * N_) / 256, 256, 0, stream>>>(vs, vsh);
    sig_kernel<<<dim3(N_ / 256, N_, B_), 256, 0, stream>>>(y1, y2, bs, sT);
    gemm_kernel<<<dim3(N_ / 256, N_ / 128, B_), 256, 0, stream>>>(vsh, sT, out);
    softmax_kernel<<<B_ * N_, 256, 0, stream>>>(out);
}
